// CausalSelfAttention_10900626997303
// MI455X (gfx1250) — matrix operand order repaired and checked by reading, not compiled, not run on hardware
//
#include <hip/hip_runtime.h>

// ---------------------------------------------------------------------------
// CDNA5 (gfx1250) fused causal self-attention pipeline, bf16 WMMA + fp32 accum
//   B=4, T=2048, C=1024, H=16, D=64
// Stages (all in d_ws):
//   0) fp32 -> bf16 conversions of x, W_in, W_out
//   1) QKV GEMM (bf16 WMMA 16x16x32) + bias + fused RoPE; K stored transposed
//   2) flash attention: block = 4 waves = 128 q rows of one (b,h);
//      K/V chunks staged in LDS via async global->LDS copies (double buffer),
//      QK^T WMMA, fp32 online softmax, P@V WMMA
//   3) output projection GEMM + bias -> fp32 d_out
// ---------------------------------------------------------------------------

#define USE_ASYNC_LDS 1  // global_load_async_to_lds_b128 + s_wait_asynccnt

typedef __attribute__((ext_vector_type(8)))  __bf16 v8bf;
typedef __attribute__((ext_vector_type(16))) __bf16 v16bf;
typedef __attribute__((ext_vector_type(8)))  float  v8f;

namespace cfg {
constexpr int kB = 4;
constexpr int kT = 2048;
constexpr int kC = 1024;
constexpr int kH = 16;
constexpr int kD = 64;
constexpr int kM = kB * kT;  // 8192 rows
constexpr int kN3 = 3 * kC;  // 3072 qkv cols
}

static __device__ __forceinline__ unsigned short f2bf(float f) {
  // round-to-nearest-even float -> bf16
  unsigned int u = __builtin_bit_cast(unsigned int, f);
  u += 0x7fffu + ((u >> 16) & 1u);
  return (unsigned short)(u >> 16);
}

static __device__ __forceinline__ v8f v8f_zero() {
  v8f z = {0.f, 0.f, 0.f, 0.f, 0.f, 0.f, 0.f, 0.f};
  return z;
}

static __device__ __forceinline__ v16bf cat16(v8bf lo, v8bf hi) {
  return __builtin_shufflevector(lo, hi, 0, 1, 2, 3, 4, 5, 6, 7,
                                 8, 9, 10, 11, 12, 13, 14, 15);
}

static __device__ __forceinline__ v8f wmma_bf16(v16bf a, v16bf b, v8f c) {
  // D = A(16x32 bf16) * B(32x16 bf16) + C(16x16 f32)
  return __builtin_amdgcn_wmma_f32_16x16x32_bf16(
      /*neg_a=*/false, a, /*neg_b=*/false, b,
      /*c_mod=*/(short)0, c, /*reuse_a=*/false, /*reuse_b=*/false);
}

// A-fragment (16x32, bf16, row-major source with contiguous K):
//   lanes 0-15  : row = lane,    elems = K[0..7]   then K[16..23]
//   lanes 16-31 : row = lane-16, elems = K[8..15]  then K[24..31]
static __device__ __forceinline__ v16bf load_afrag(const __bf16* row_base,
                                                   int kbase_lo_hi) {
  v8bf lo = *(const v8bf*)(row_base + kbase_lo_hi);
  v8bf hi = *(const v8bf*)(row_base + kbase_lo_hi + 16);
  return cat16(lo, hi);
}

// B-fragment (32x16, bf16): lane%16 = column N, elems = K[8h..8h+7] then
// K[16+8h..16+8h+7] (p points at K=8h, stride ld between K rows).
static __device__ __forceinline__ v16bf load_bfrag(const __bf16* p, int ld) {
  v16bf r;
#pragma unroll
  for (int s = 0; s < 8; ++s) {
    r[s] = p[(size_t)s * ld];
    r[s + 8] = p[(size_t)(16 + s) * ld];
  }
  return r;
}

// 16B global -> LDS copy. Async path uses the CDNA5 async data mover
// (tracked with ASYNCcnt); fallback is a plain load + ds_store.
static __device__ __forceinline__ void cp_g2l_b128(const __bf16* g,
                                                   __bf16* lds) {
#if USE_ASYNC_LDS
  unsigned lds_off = (unsigned)(unsigned long long)lds;  // low 32b = LDS addr
  asm volatile("global_load_async_to_lds_b128 %0, %1, off"
               :
               : "v"(lds_off), "v"(g)
               : "memory");
#else
  *(v8bf*)lds = *(const v8bf*)g;
#endif
}

static __device__ __forceinline__ void wait_async_copies() {
#if USE_ASYNC_LDS
  asm volatile("s_wait_asynccnt 0x0" ::: "memory");
#endif
}

// ---------------------------------------------------------------------------
// Stage 0: fp32 -> bf16
// ---------------------------------------------------------------------------
__global__ void __launch_bounds__(256) convert_f32_bf16(
    const float* __restrict__ in, unsigned short* __restrict__ out, int n) {
  int i = blockIdx.x * 256 + threadIdx.x;
  if (i < n) out[i] = f2bf(in[i]);
}

// ---------------------------------------------------------------------------
// Stage 1: qkv = x @ W_in + b_in ; split; RoPE(q,k); scatter to workspaces.
//   grid = (M/64, 3C/64), block = 128 (4 waves); wave w owns 16x64 tile.
//   q  -> [B,H,T,D] bf16, k -> [B,H,D,T] bf16 (transposed!), v -> [B,H,T,D]
// ---------------------------------------------------------------------------
__global__ void __launch_bounds__(128) qkv_rope_kernel(
    const __bf16* __restrict__ xb,     // [M, C]
    const __bf16* __restrict__ wib,    // [C, 3C]
    const float* __restrict__ b_in,    // [3C]
    unsigned short* __restrict__ qg,   // [B,H,T,D]
    unsigned short* __restrict__ ktg,  // [B,H,D,T]
    unsigned short* __restrict__ vg)   // [B,H,T,D]
{
  using namespace cfg;
  const int lane = threadIdx.x & 31;
  const int wave = threadIdx.x >> 5;
  const int lm = lane & 15;
  const bool hiH = lane >= 16;
  const int m0 = blockIdx.x * 64 + wave * 16;
  const int n0 = blockIdx.y * 64;

  v8f acc[4];
#pragma unroll
  for (int t = 0; t < 4; ++t) acc[t] = v8f_zero();

  const __bf16* arow = xb + (size_t)(m0 + lm) * kC;
  for (int kk = 0; kk < kC; kk += 32) {
    const int ka = kk + (hiH ? 8 : 0);
    __builtin_prefetch(arow + ka + 64, 0, 1);  // global_prefetch_b8
    v16bf a = load_afrag(arow, ka);
    const __bf16* brow = wib + (size_t)(kk + (hiH ? 8 : 0)) * kN3 + n0 + lm;
#pragma unroll
    for (int t = 0; t < 4; ++t)
      acc[t] = wmma_bf16(a, load_bfrag(brow + 16 * t, kN3), acc[t]);
  }

  // epilogue: bias + RoPE + scatter. C/D layout: VGPR i -> row i (+8 upper
  // half), lane%16 -> column.
#pragma unroll
  for (int t = 0; t < 4; ++t) {
    const int n = n0 + 16 * t + lm;  // [0, 3C)
    const int region = n >> 10;      // 0=Q 1=K 2=V (uniform across wave)
    const int nc = n & (kC - 1);
    const int h = nc >> 6;
    const int d = nc & (kD - 1);
    const float bias = b_in[n];
#pragma unroll
    for (int i = 0; i < 8; ++i) {
      const int m = m0 + i + (hiH ? 8 : 0);
      const int tok = m & (kT - 1);
      const int bb = m >> 11;  // m / T
      float v = acc[t][i] + bias;
      if (region < 2) {
        // RoPE: even/odd dim pair lives in adjacent lanes of the same row.
        float partner = __shfl_xor(v, 1, 32);
        const int ip = d >> 1;
        const float inv_freq =
            __expf(-(float)(2 * ip) * (9.210340371976184f / (float)kD));
        const float theta = (float)tok * inv_freq;
        const float c = __cosf(theta);
        const float s = __sinf(theta);
        v = (d & 1) ? (v * c + partner * s) : (v * c - partner * s);
      }
      const size_t bh = (size_t)bb * kH + h;
      const unsigned short bits = f2bf(v);
      if (region == 0) {
        qg[(bh * kT + tok) * kD + d] = bits;
      } else if (region == 1) {
        ktg[(bh * kD + d) * kT + tok] = bits;  // transposed for QK^T B-frags
      } else {
        vg[(bh * kT + tok) * kD + d] = bits;
      }
    }
  }
}

// ---------------------------------------------------------------------------
// Stage 2: flash attention.
//   Block = 128 threads (4 waves) owns 128 consecutive q rows of one (b,h);
//   wave w owns rows [q0+32w, q0+32w+32) as two 16-row m-tiles.
//   Per 32-key chunk, K^T (64d x 32k) and V (32k x 64d) are staged once per
//   block into double-buffered LDS via async global->LDS copies; every
//   LDS B-fragment then feeds two WMMAs (two m-tiles).
// ---------------------------------------------------------------------------
static __device__ __forceinline__ void issue_chunk_copy(
    const __bf16* Kt, const __bf16* Vb, int kb, __bf16 (*Ktile)[32],
    __bf16 (*Vtile)[64], int tid) {
  using namespace cfg;
  // K^T chunk: 64 rows(d) x 32 keys = 64 x 64B = 256 x 16B segments
#pragma unroll
  for (int r = 0; r < 2; ++r) {
    const int seg = tid * 2 + r;
    const int d = seg >> 2;
    const int part = seg & 3;
    cp_g2l_b128(Kt + (size_t)d * kT + kb + part * 8, &Ktile[d][part * 8]);
  }
  // V chunk: 32 rows(key) x 64 d = 32 x 128B = 256 x 16B segments
#pragma unroll
  for (int r = 0; r < 2; ++r) {
    const int seg = tid * 2 + r;
    const int key = seg >> 3;
    const int part = seg & 7;
    cp_g2l_b128(Vb + (size_t)(kb + key) * kD + part * 8,
                &Vtile[key][part * 8]);
  }
}

__global__ void __launch_bounds__(128) flash_attn_kernel(
    const __bf16* __restrict__ qg,    // [B,H,T,D]
    const __bf16* __restrict__ ktg,   // [B,H,D,T]
    const __bf16* __restrict__ vg,    // [B,H,T,D]
    unsigned short* __restrict__ yg)  // [B,T,C] bf16
{
  using namespace cfg;
  __shared__ alignas(16) __bf16 Ktile[2][64][32];   // 8 KB  (double buffer)
  __shared__ alignas(16) __bf16 Vtile[2][32][64];   // 8 KB
  __shared__ alignas(16) __bf16 Pbuf[4][2][16][32]; // 8 KB  (per wave/m-tile)

  const int tid = threadIdx.x;
  const int lane = tid & 31;
  const int wave = tid >> 5;
  const int lm = lane & 15;
  const bool hiH = lane >= 16;

  constexpr int kBlocksPerT = kT / 128;  // 16
  const int q0 = (blockIdx.x & (kBlocksPerT - 1)) * 128;
  const int h = (blockIdx.x / kBlocksPerT) & (kH - 1);
  const int b = blockIdx.x / (kBlocksPerT * kH);

  const size_t bh = (size_t)b * kH + h;
  const __bf16* Qb = qg + bh * (size_t)(kT * kD);
  const __bf16* Kt = ktg + bh * (size_t)(kD * kT);
  const __bf16* Vb = vg + bh * (size_t)(kT * kD);

  const int qw = q0 + wave * 32;  // this wave's 32 q rows

  // Q fragments: 2 m-tiles x (16 rows x 64 dims) = 2 x two 16x32 A-fragments
  v16bf aq[2][2];
#pragma unroll
  for (int mt = 0; mt < 2; ++mt) {
    const __bf16* qrow = Qb + (size_t)(qw + mt * 16 + lm) * kD;
#pragma unroll
    for (int c2 = 0; c2 < 2; ++c2)
      aq[mt][c2] = load_afrag(qrow, c2 * 32 + (hiH ? 8 : 0));
  }

  v8f o[2][4];
  float mrow[2][8], lrow[2][8];
#pragma unroll
  for (int mt = 0; mt < 2; ++mt) {
#pragma unroll
    for (int t = 0; t < 4; ++t) o[mt][t] = v8f_zero();
#pragma unroll
    for (int i = 0; i < 8; ++i) {
      mrow[mt][i] = -__builtin_inff();
      lrow[mt][i] = 0.f;
    }
  }

  const float scale = 0.125f;                // 1/sqrt(64)
  const int nchunks = (q0 + 128 + 31) >> 5;  // block-uniform causal bound

  issue_chunk_copy(Kt, Vb, 0, Ktile[0], Vtile[0], tid);

  for (int ch = 0; ch < nchunks; ++ch) {
    const int kb = ch * 32;
    const int buf = ch & 1;

    wait_async_copies();  // my copies for buf done
    __syncthreads();      // everyone's copies visible; old reads retired
    if (ch + 1 < nchunks)
      issue_chunk_copy(Kt, Vb, kb + 32, Ktile[buf ^ 1], Vtile[buf ^ 1], tid);

    if (kb < qw + 32) {  // wave-uniform: chunk intersects this wave's rows
      // --- K^T B-fragments from LDS (lane%16 = key, elems = d) ---
      v16bf bk[2][2];  // [key-tile][d-chunk]
#pragma unroll
      for (int tile = 0; tile < 2; ++tile)
#pragma unroll
        for (int c2 = 0; c2 < 2; ++c2)
          bk[tile][c2] = load_bfrag(&Ktile[buf][c2 * 32 + (hiH ? 8 : 0)][tile * 16 + lm], 32);

      // --- S = Q K^T : 2 m-tiles x 2 key-tiles, contraction over d=64 ---
      v8f s[2][2];
#pragma unroll
      for (int mt = 0; mt < 2; ++mt)
#pragma unroll
        for (int tile = 0; tile < 2; ++tile) {
          v8f acc = v8f_zero();
          acc = wmma_bf16(aq[mt][0], bk[tile][0], acc);
          acc = wmma_bf16(aq[mt][1], bk[tile][1], acc);
          s[mt][tile] = acc;
        }

      // --- fp32 online softmax (row stats per C-layout row) ---
      float alpha[2][8];
#pragma unroll
      for (int mt = 0; mt < 2; ++mt)
#pragma unroll
        for (int i = 0; i < 8; ++i) {
          const int row = qw + mt * 16 + i + (hiH ? 8 : 0);
          float s0 = s[mt][0][i] * scale;
          float s1 = s[mt][1][i] * scale;
          s0 = (kb + lm > row) ? -__builtin_inff() : s0;       // causal mask
          s1 = (kb + 16 + lm > row) ? -__builtin_inff() : s1;  // (selects)
          float mx = fmaxf(s0, s1);
          mx = fmaxf(mx, __shfl_xor(mx, 1, 32));
          mx = fmaxf(mx, __shfl_xor(mx, 2, 32));
          mx = fmaxf(mx, __shfl_xor(mx, 4, 32));
          mx = fmaxf(mx, __shfl_xor(mx, 8, 32));  // row-max in 16-lane half
          const float mnew = fmaxf(mrow[mt][i], mx);
          const float a = __expf(mrow[mt][i] - mnew);
          const float p0 = __expf(s0 - mnew);
          const float p1 = __expf(s1 - mnew);
          float sum = p0 + p1;
          sum += __shfl_xor(sum, 1, 32);
          sum += __shfl_xor(sum, 2, 32);
          sum += __shfl_xor(sum, 4, 32);
          sum += __shfl_xor(sum, 8, 32);
          lrow[mt][i] = lrow[mt][i] * a + sum;
          mrow[mt][i] = mnew;
          alpha[mt][i] = a;
          // stage P (bf16) into per-wave LDS, plain row-major 16x32
          const int r = i + (hiH ? 8 : 0);
          Pbuf[wave][mt][r][lm] = __builtin_bit_cast(__bf16, f2bf(p0));
          Pbuf[wave][mt][r][lm + 16] = __builtin_bit_cast(__bf16, f2bf(p1));
        }
      asm volatile("s_wait_dscnt 0x0" ::: "memory");  // LDS RAW within wave

      // reload P as 16x32 A-fragments
      v16bf ap[2];
#pragma unroll
      for (int mt = 0; mt < 2; ++mt)
        ap[mt] = load_afrag(&Pbuf[wave][mt][lm][0], hiH ? 8 : 0);

      // V B-fragments from LDS (lane%16 = d column, elems = key)
      v16bf bv[4];
#pragma unroll
      for (int t = 0; t < 4; ++t)
        bv[t] = load_bfrag(&Vtile[buf][hiH ? 8 : 0][t * 16 + lm], 64);

      // rescale running O, then O += P @ V (4 d-tiles of 16)
#pragma unroll
      for (int mt = 0; mt < 2; ++mt) {
#pragma unroll
        for (int t = 0; t < 4; ++t)
#pragma unroll
          for (int i = 0; i < 8; ++i) o[mt][t][i] *= alpha[mt][i];
#pragma unroll
        for (int t = 0; t < 4; ++t)
          o[mt][t] = wmma_bf16(ap[mt], bv[t], o[mt][t]);
      }
    }
  }

  // normalize and write y[B,T,C] bf16
#pragma unroll
  for (int mt = 0; mt < 2; ++mt)
#pragma unroll
    for (int t = 0; t < 4; ++t)
#pragma unroll
      for (int i = 0; i < 8; ++i) {
        const int row = qw + mt * 16 + i + (hiH ? 8 : 0);
        const float v = o[mt][t][i] / lrow[mt][i];
        yg[((size_t)b * kT + row) * kC + h * kD + t * 16 + lm] = f2bf(v);
      }
}

// ---------------------------------------------------------------------------
// Stage 3: out = y @ W_out + b_out (fp32 result)
//   grid = (M/64, C/64), block = 128 (4 waves).
// ---------------------------------------------------------------------------
__global__ void __launch_bounds__(128) out_proj_kernel(
    const __bf16* __restrict__ yb,    // [M, C]
    const __bf16* __restrict__ wob,   // [C, C]
    const float* __restrict__ b_out,  // [C]
    float* __restrict__ out)          // [M, C]
{
  using namespace cfg;
  const int lane = threadIdx.x & 31;
  const int wave = threadIdx.x >> 5;
  const int lm = lane & 15;
  const bool hiH = lane >= 16;
  const int m0 = blockIdx.x * 64 + wave * 16;
  const int n0 = blockIdx.y * 64;

  v8f acc[4];
#pragma unroll
  for (int t = 0; t < 4; ++t) acc[t] = v8f_zero();

  const __bf16* arow = yb + (size_t)(m0 + lm) * kC;
  for (int kk = 0; kk < kC; kk += 32) {
    const int ka = kk + (hiH ? 8 : 0);
    __builtin_prefetch(arow + ka + 64, 0, 1);
    v16bf a = load_afrag(arow, ka);
    const __bf16* brow = wob + (size_t)(kk + (hiH ? 8 : 0)) * kC + n0 + lm;
#pragma unroll
    for (int t = 0; t < 4; ++t)
      acc[t] = wmma_bf16(a, load_bfrag(brow + 16 * t, kC), acc[t]);
  }

#pragma unroll
  for (int t = 0; t < 4; ++t) {
    const int n = n0 + 16 * t + lm;
    const float bias = b_out[n];
#pragma unroll
    for (int i = 0; i < 8; ++i) {
      const int m = m0 + i + (hiH ? 8 : 0);
      out[(size_t)m * kC + n] = acc[t][i] + bias;
    }
  }
}

// ---------------------------------------------------------------------------
// Host-side launcher (graph-capture safe: no alloc/sync, only stream work).
// ---------------------------------------------------------------------------
extern "C" void kernel_launch(void* const* d_in, const int* in_sizes, int n_in,
                              void* d_out, int out_size, void* d_ws,
                              size_t ws_size, hipStream_t stream) {
  using namespace cfg;
  const float* x = (const float*)d_in[0];      // [B,T,C]
  const float* W_in = (const float*)d_in[1];   // [C,3C]
  const float* b_in = (const float*)d_in[2];   // [3C]
  const float* W_out = (const float*)d_in[3];  // [C,C]
  const float* b_out = (const float*)d_in[4];  // [C]
  float* out = (float*)d_out;                  // [B,T,C] fp32

  // Workspace carve-out (bf16 elements); total ~= 92.3 MB.
  unsigned short* ws = (unsigned short*)d_ws;
  size_t off = 0;
  auto carve = [&](size_t n) {
    unsigned short* p = ws + off;
    off += n;
    return p;
  };
  const size_t nX = (size_t)kM * kC;  // 8.39M
  unsigned short* xb = carve(nX);
  unsigned short* wib = carve((size_t)kC * kN3);
  unsigned short* wob = carve((size_t)kC * kC);
  unsigned short* qg = carve(nX);   // [B,H,T,D]
  unsigned short* ktg = carve(nX);  // [B,H,D,T]
  unsigned short* vg = carve(nX);   // [B,H,T,D]
  unsigned short* yg = carve(nX);   // [B,T,C]
  (void)ws_size;
  (void)in_sizes;
  (void)n_in;
  (void)out_size;

  // Stage 0: conversions
  convert_f32_bf16<<<(int)((nX + 255) / 256), 256, 0, stream>>>(x, xb, (int)nX);
  convert_f32_bf16<<<(kC * kN3 + 255) / 256, 256, 0, stream>>>(W_in, wib,
                                                               kC * kN3);
  convert_f32_bf16<<<(kC * kC + 255) / 256, 256, 0, stream>>>(W_out, wob,
                                                              kC * kC);

  // Stage 1: QKV + RoPE
  qkv_rope_kernel<<<dim3(kM / 64, kN3 / 64), 128, 0, stream>>>(
      (const __bf16*)xb, (const __bf16*)wib, b_in, qg, ktg, vg);

  // Stage 2: flash attention (one block per 128 q-rows of one (b,h))
  flash_attn_kernel<<<kB * kH * (kT / 128), 128, 0, stream>>>(
      (const __bf16*)qg, (const __bf16*)ktg, (const __bf16*)vg, yg);

  // Stage 3: output projection -> fp32
  out_proj_kernel<<<dim3(kM / 64, kC / 64), 128, 0, stream>>>(
      (const __bf16*)yg, (const __bf16*)wob, b_out, out);
}
